// SamplingAttentionDecoder_57690000719932
// MI455X (gfx1250) — compile-verified, matrix operand order checked by reading, not run on hardware
//
#include <hip/hip_runtime.h>

typedef unsigned short ushort_t;
typedef unsigned int uint32;
typedef __attribute__((ext_vector_type(16))) __bf16 v16bf;
typedef __attribute__((ext_vector_type(8)))  float  v8f;
typedef __attribute__((ext_vector_type(4)))  int    v4i;
typedef __attribute__((address_space(1))) v4i* gas_v4i_ptr;   // global b128 chunk
typedef __attribute__((address_space(3))) v4i* lds_v4i_ptr;   // LDS b128 chunk

#define B_    32
#define TDEC  64
#define TFEAT 512
#define MDIM  512
#define VOC   16000
#define ATT_SCALE 0.044194173824159216f  // 1/sqrt(512)

#if __has_builtin(__builtin_amdgcn_global_load_async_to_lds_b128)
#define HAVE_ASYNC_LDS 1
#endif

// ---------- helpers ----------
__device__ __forceinline__ ushort_t f2bf(float f) {
    uint32 u = __builtin_bit_cast(uint32, f);
    uint32 r = u + 0x7fffu + ((u >> 16) & 1u);   // round-to-nearest-even
    return (ushort_t)(r >> 16);
}
__device__ __forceinline__ __bf16 bfc(ushort_t u) {
    return __builtin_bit_cast(__bf16, u);
}
__device__ __forceinline__ float hash_uniform(uint32 a) {
    a ^= a >> 16; a *= 0x7feb352du; a ^= a >> 15; a *= 0x846ca68bu; a ^= a >> 16;
    return (float)(a >> 8) * (1.0f / 16777216.0f);
}

// Load one WMMA 16-bit operand tile fragment for this lane.
// p points at (row, k0) of a row-major bf16 matrix; lane covers
// K = hi*8..hi*8+7 and K = 16+hi*8..16+hi*8+7 (two 16-byte segments).
__device__ __forceinline__ v16bf load_frag(const ushort_t* p, int hi) {
    v16bf r;
    const uint32* q0 = (const uint32*)(p + hi * 8);
    const uint32* q1 = (const uint32*)(p + 16 + hi * 8);
    #pragma unroll
    for (int v = 0; v < 4; ++v) {
        uint32 w = q0[v];
        r[2 * v]     = bfc((ushort_t)(w & 0xffffu));
        r[2 * v + 1] = bfc((ushort_t)(w >> 16));
    }
    #pragma unroll
    for (int v = 0; v < 4; ++v) {
        uint32 w = q1[v];
        r[8 + 2 * v]     = bfc((ushort_t)(w & 0xffffu));
        r[8 + 2 * v + 1] = bfc((ushort_t)(w >> 16));
    }
    return r;
}

// ---------- fp32 -> bf16 conversion ----------
__global__ void f32_to_bf16(const float* __restrict__ src, ushort_t* __restrict__ dst, int n) {
    int i = blockIdx.x * blockDim.x + threadIdx.x;
    if (i < n) dst[i] = f2bf(src[i]);
}

// ---------- bf16 WMMA GEMM:  C[Mr,N] = A[Mr,K] @ W[N,K]^T + bias ----------
// block = 256 threads (8 waves); grid = (N/128, Mr/(16*MT)).
// The A strip [MT*16, K] is copied once into LDS (async-to-LDS path), then the
// K loop runs barrier-free, ping-pong unrolled by 128 so the prefetched B tiles
// are consumed in place (no register-copy double-buffer shuffle).
template <int MT>
__global__ void gemm_bf16_wmma(const ushort_t* __restrict__ A,
                               const ushort_t* __restrict__ W,
                               const float* __restrict__ bias,
                               float* __restrict__ C,
                               int Mr, int N, int K, int ldc)
{
    extern __shared__ ushort_t As[];          // [MT*16][K] row-major, set at launch
    const int lane  = threadIdx.x & 31;
    const int wave  = threadIdx.x >> 5;
    const int n0    = (blockIdx.x * 8 + wave) * 16;
    const int m0    = blockIdx.y * (16 * MT);
    const int mlane = lane & 15;              // A: M index / B: N index
    const int hi    = lane >> 4;              // K half-group for this lane

    // ---- stage A strip into LDS, 16 bytes (8 bf16) per transfer ----
    const int segs = K >> 3;                  // 16B segments per row
    const int tot  = MT * 16 * segs;          // multiple of 256 for K in {512,1024}
#ifdef HAVE_ASYNC_LDS
    for (int i = threadIdx.x; i < tot; i += 256) {
        int row = i / segs, seg = i - row * segs;
        const ushort_t* g = A + (size_t)(m0 + row) * K + (size_t)seg * 8;
        ushort_t* l = As + (size_t)row * K + (size_t)seg * 8;
        __builtin_amdgcn_global_load_async_to_lds_b128(
            (gas_v4i_ptr)g, (lds_v4i_ptr)l, 0, 0);
    }
#if __has_builtin(__builtin_amdgcn_s_wait_asynccnt)
    __builtin_amdgcn_s_wait_asynccnt(0);
#else
    asm volatile("s_wait_asynccnt 0x0" ::: "memory");
#endif
#else
    for (int i = threadIdx.x; i < tot; i += 256) {
        int row = i / segs, seg = i - row * segs;
        const uint4* g = (const uint4*)(A + (size_t)(m0 + row) * K + (size_t)seg * 8);
        *(uint4*)((char*)As + ((size_t)row * K + seg * 8) * 2) = *g;
    }
#endif
    __syncthreads();

    const ushort_t* wrow = W + (size_t)(n0 + mlane) * K;
    v8f acc[MT];
    #pragma unroll
    for (int mt = 0; mt < MT; ++mt) acc[mt] = (v8f){};

    // ping-pong B stream; K is a multiple of 128
    v16bf bA0 = load_frag(wrow, hi);
    v16bf bA1 = load_frag(wrow + 32, hi);
    for (int k0 = 0; k0 < K; k0 += 128) {
        v16bf bB0 = load_frag(wrow + k0 + 64, hi);
        v16bf bB1 = load_frag(wrow + k0 + 96, hi);
        #pragma unroll
        for (int mt = 0; mt < MT; ++mt) {
            v16bf a = load_frag(&As[(size_t)(mt * 16 + mlane) * K + k0], hi);
            acc[mt] = __builtin_amdgcn_wmma_f32_16x16x32_bf16(false, a, false, bA0,
                                                              (short)0, acc[mt], false, false);
        }
        #pragma unroll
        for (int mt = 0; mt < MT; ++mt) {
            v16bf a = load_frag(&As[(size_t)(mt * 16 + mlane) * K + k0 + 32], hi);
            acc[mt] = __builtin_amdgcn_wmma_f32_16x16x32_bf16(false, a, false, bA1,
                                                              (short)0, acc[mt], false, false);
        }
        if (k0 + 128 < K) {
            bA0 = load_frag(wrow + k0 + 128, hi);
            bA1 = load_frag(wrow + k0 + 160, hi);
        }
        #pragma unroll
        for (int mt = 0; mt < MT; ++mt) {
            v16bf a = load_frag(&As[(size_t)(mt * 16 + mlane) * K + k0 + 64], hi);
            acc[mt] = __builtin_amdgcn_wmma_f32_16x16x32_bf16(false, a, false, bB0,
                                                              (short)0, acc[mt], false, false);
        }
        #pragma unroll
        for (int mt = 0; mt < MT; ++mt) {
            v16bf a = load_frag(&As[(size_t)(mt * 16 + mlane) * K + k0 + 96], hi);
            acc[mt] = __builtin_amdgcn_wmma_f32_16x16x32_bf16(false, a, false, bB1,
                                                              (short)0, acc[mt], false, false);
        }
    }

    const int n  = n0 + mlane;
    const float bv = bias ? bias[n] : 0.0f;
    #pragma unroll
    for (int mt = 0; mt < MT; ++mt) {
        #pragma unroll
        for (int r = 0; r < 8; ++r) {
            int m = m0 + mt * 16 + hi * 8 + r;   // C/D layout: lanes 16-31 hold M=8..15
            C[(size_t)m * ldc + n] = acc[mt][r] + bv;
        }
    }
}

// ---------- categorical sample + build x = [emb[idx], context] in bf16 ----------
__global__ void sample_embed(const int* __restrict__ toks,
                             const float* __restrict__ logits_base, // outs [B,TDEC,VOC]
                             const float* __restrict__ emb,
                             const float* __restrict__ context,
                             ushort_t* __restrict__ x_bf, int t)
{
    __shared__ float part[256];
    __shared__ float s_gmax, s_Z;
    __shared__ int   sidx;
    const int b = blockIdx.x, tid = threadIdx.x;
    const int tok = toks[b * TDEC + t];
    if (tid == 0) sidx = tok;          // t==0: pure one-hot -> tok; also fp fallback
    __syncthreads();

    if (t > 0) {
        const float* lg = logits_base + (size_t)b * TDEC * VOC + (size_t)(t - 1) * VOC;
        const int CH = (VOC + 255) / 256;   // 63
        int v0 = tid * CH, v1 = v0 + CH; if (v1 > VOC) v1 = VOC;

        float lm = -1e30f;
        for (int v = v0; v < v1; ++v) lm = fmaxf(lm, lg[v]);
        part[tid] = lm; __syncthreads();
        for (int s = 128; s > 0; s >>= 1) { if (tid < s) part[tid] = fmaxf(part[tid], part[tid + s]); __syncthreads(); }
        if (tid == 0) s_gmax = part[0];
        __syncthreads();
        const float gmax = s_gmax;

        float ls = 0.f;
        for (int v = v0; v < v1; ++v) ls += __expf(lg[v] - gmax);
        part[tid] = ls; __syncthreads();
        for (int s = 128; s > 0; s >>= 1) { if (tid < s) part[tid] += part[tid + s]; __syncthreads(); }
        if (tid == 0) s_Z = part[0];
        __syncthreads();
        const float Z = s_Z;

        // mixed dist: 0.5*softmax + 0.5*onehot(tok); chunk masses -> exclusive prefix
        float mass = 0.5f * ls / Z + ((tok >= v0 && tok < v1) ? 0.5f : 0.f);
        part[tid] = mass; __syncthreads();
        if (tid == 0) { float c = 0.f; for (int i = 0; i < 256; ++i) { float m = part[i]; part[i] = c; c += m; } }
        __syncthreads();

        const float u = hash_uniform((uint32)(t * 9781u + b * 6271u + 0x9e3779b9u));
        const float pref = part[tid];
        if (u >= pref && u < pref + mass) {
            float c = pref;
            for (int v = v0; v < v1; ++v) {
                c += 0.5f * __expf(lg[v] - gmax) / Z + ((v == tok) ? 0.5f : 0.f);
                if (u < c) { sidx = v; break; }
            }
        }
        __syncthreads();
    }

    const int idx = sidx;
    const float* erow = emb + (size_t)idx * MDIM;
    const float* ctx  = context + (size_t)b * MDIM;
    for (int m = tid; m < MDIM; m += 256) {
        x_bf[b * 2 * MDIM + m]        = f2bf(erow[m]);
        x_bf[b * 2 * MDIM + MDIM + m] = f2bf(ctx[m]);
    }
}

// ---------- GRU gate fusion ----------
__global__ void gru_update(const float* __restrict__ gx, const float* __restrict__ gh,
                           float* __restrict__ hidden, ushort_t* __restrict__ hidden_bf)
{
    int i = blockIdx.x * blockDim.x + threadIdx.x;
    if (i >= B_ * MDIM) return;
    int b = i / MDIM, m = i - b * MDIM;
    const float* gxr = gx + (size_t)b * 3 * MDIM;
    const float* ghr = gh + (size_t)b * 3 * MDIM;
    float r = 1.f / (1.f + __expf(-(gxr[m] + ghr[m])));
    float z = 1.f / (1.f + __expf(-(gxr[MDIM + m] + ghr[MDIM + m])));
    float n = tanhf(gxr[2 * MDIM + m] + r * ghr[2 * MDIM + m]);
    float h = (1.f - z) * n + z * hidden[i];
    hidden[i] = h;
    hidden_bf[i] = f2bf(h);
}

// ---------- attention: scores/softmax/context + emit attw and hc_bf ----------
__global__ void attention_kernel(const float* __restrict__ qv, const float* __restrict__ Kf,
                                 const float* __restrict__ Vf, const unsigned char* __restrict__ fmask,
                                 const float* __restrict__ hidden,
                                 float* __restrict__ context, ushort_t* __restrict__ hc_bf,
                                 float* __restrict__ attw, int t)
{
    __shared__ float qs[MDIM];
    __shared__ float sc[TFEAT];
    __shared__ float part[256];
    __shared__ float s_gmax, s_Z;
    const int b = blockIdx.x, tid = threadIdx.x;

    for (int m = tid; m < MDIM; m += 256) qs[m] = qv[b * MDIM + m];
    __syncthreads();

    for (int tf = tid; tf < TFEAT; tf += 256) {
        const float* kr = Kf + ((size_t)b * TFEAT + tf) * MDIM;
        float s = 0.f;
        #pragma unroll 4
        for (int m = 0; m < MDIM; ++m) s += qs[m] * kr[m];
        s *= ATT_SCALE;
        if (!fmask[b * TFEAT + tf]) s = -1e9f;
        sc[tf] = s;
    }
    __syncthreads();

    float lm = -1e30f;
    for (int tf = tid; tf < TFEAT; tf += 256) lm = fmaxf(lm, sc[tf]);
    part[tid] = lm; __syncthreads();
    for (int s = 128; s > 0; s >>= 1) { if (tid < s) part[tid] = fmaxf(part[tid], part[tid + s]); __syncthreads(); }
    if (tid == 0) s_gmax = part[0];
    __syncthreads();
    const float gmax = s_gmax;

    float ls = 0.f;
    for (int tf = tid; tf < TFEAT; tf += 256) { float e = __expf(sc[tf] - gmax); sc[tf] = e; ls += e; }
    part[tid] = ls; __syncthreads();
    for (int s = 128; s > 0; s >>= 1) { if (tid < s) part[tid] += part[tid + s]; __syncthreads(); }
    if (tid == 0) s_Z = part[0];
    __syncthreads();
    const float inv = 1.f / s_Z;

    for (int tf = tid; tf < TFEAT; tf += 256) {
        float w = sc[tf] * inv;
        sc[tf] = w;
        attw[((size_t)b * TFEAT + tf) * TDEC + t] = w;   // ws layout [B, T_feat, T_dec]
    }
    __syncthreads();

    for (int m = tid; m < MDIM; m += 256) {
        float acc = 0.f;
        for (int tf = 0; tf < TFEAT; ++tf) acc += sc[tf] * Vf[((size_t)b * TFEAT + tf) * MDIM + m];
        context[b * MDIM + m] = acc;
        hc_bf[b * 2 * MDIM + MDIM + m] = f2bf(acc);
        hc_bf[b * 2 * MDIM + m]        = f2bf(hidden[b * MDIM + m]);
    }
}

// ---------- host ----------
extern "C" void kernel_launch(void* const* d_in, const int* in_sizes, int n_in,
                              void* d_out, int out_size, void* d_ws, size_t ws_size,
                              hipStream_t stream)
{
    const int*   input    = (const int*)  d_in[0];
    const float* features = (const float*)d_in[1];
    const unsigned char* fmask = (const unsigned char*)d_in[2];
    const float* emb  = (const float*)d_in[3];
    const float* w_ih = (const float*)d_in[4];
    const float* w_hh = (const float*)d_in[5];
    const float* b_ih = (const float*)d_in[6];
    const float* b_hh = (const float*)d_in[7];
    const float* q_w  = (const float*)d_in[8];
    const float* q_b  = (const float*)d_in[9];
    const float* k_w  = (const float*)d_in[10];
    const float* k_b  = (const float*)d_in[11];
    const float* v_w  = (const float*)d_in[12];
    const float* v_b  = (const float*)d_in[13];
    const float* out_w = (const float*)d_in[14];
    const float* out_b = (const float*)d_in[15];

    float* outs = (float*)d_out;                       // [B, TDEC, VOC]
    float* attw = outs + (size_t)B_ * TDEC * VOC;      // [B, TFEAT, TDEC]

    char* ws = (char*)d_ws;
    size_t off = 0;
    auto alloc = [&](size_t bytes) -> char* {
        char* p = ws + off; off = (off + bytes + 255) & ~(size_t)255; return p;
    };
    ushort_t* w_ih_bf  = (ushort_t*)alloc((size_t)3 * MDIM * 2 * MDIM * 2);
    ushort_t* w_hh_bf  = (ushort_t*)alloc((size_t)3 * MDIM * MDIM * 2);
    ushort_t* q_w_bf   = (ushort_t*)alloc((size_t)MDIM * MDIM * 2);
    ushort_t* k_w_bf   = (ushort_t*)alloc((size_t)MDIM * MDIM * 2);
    ushort_t* v_w_bf   = (ushort_t*)alloc((size_t)MDIM * MDIM * 2);
    ushort_t* out_w_bf = (ushort_t*)alloc((size_t)VOC * 2 * MDIM * 2);
    ushort_t* feat_bf  = (ushort_t*)alloc((size_t)B_ * TFEAT * MDIM * 2);
    float*    Kf       = (float*)   alloc((size_t)B_ * TFEAT * MDIM * 4);
    float*    Vf       = (float*)   alloc((size_t)B_ * TFEAT * MDIM * 4);
    float*    hidden   = (float*)   alloc((size_t)B_ * MDIM * 4);
    float*    context  = (float*)   alloc((size_t)B_ * MDIM * 4);
    float*    qv       = (float*)   alloc((size_t)B_ * MDIM * 4);
    float*    gx       = (float*)   alloc((size_t)B_ * 3 * MDIM * 4);
    float*    gh       = (float*)   alloc((size_t)B_ * 3 * MDIM * 4);
    ushort_t* hidden_bf = (ushort_t*)alloc((size_t)B_ * MDIM * 2);
    ushort_t* x_bf      = (ushort_t*)alloc((size_t)B_ * 2 * MDIM * 2);
    ushort_t* hc_bf     = (ushort_t*)alloc((size_t)B_ * 2 * MDIM * 2);

    auto conv = [&](const float* s, ushort_t* d, size_t n) {
        f32_to_bf16<<<dim3((unsigned)((n + 255) / 256)), dim3(256), 0, stream>>>(s, d, (int)n);
    };
    conv(w_ih,  w_ih_bf,  (size_t)3 * MDIM * 2 * MDIM);
    conv(w_hh,  w_hh_bf,  (size_t)3 * MDIM * MDIM);
    conv(q_w,   q_w_bf,   (size_t)MDIM * MDIM);
    conv(k_w,   k_w_bf,   (size_t)MDIM * MDIM);
    conv(v_w,   v_w_bf,   (size_t)MDIM * MDIM);
    conv(out_w, out_w_bf, (size_t)VOC * 2 * MDIM);
    conv(features, feat_bf, (size_t)B_ * TFEAT * MDIM);

    (void)hipMemsetAsync(hidden,    0, (size_t)B_ * MDIM * 4, stream);
    (void)hipMemsetAsync(context,   0, (size_t)B_ * MDIM * 4, stream);
    (void)hipMemsetAsync(hidden_bf, 0, (size_t)B_ * MDIM * 2, stream);

    // MT=2: 32 activation rows per block -> weight matrix streamed once from L2
    auto gemm = [&](const ushort_t* A, const ushort_t* W, const float* bias,
                    float* C, int Mr, int N, int K, int ldc) {
        size_t lds_bytes = (size_t)2 * 16 * K * 2;   // A strip [32, K] bf16
        gemm_bf16_wmma<2><<<dim3(N / 128, Mr / 32), dim3(256), lds_bytes, stream>>>(
            A, W, bias, C, Mr, N, K, ldc);
    };

    // Encoder-side projections (one-time): K = feat @ k_w^T + k_b ; V likewise
    gemm(feat_bf, k_w_bf, k_b, Kf, B_ * TFEAT, MDIM, MDIM, MDIM);
    gemm(feat_bf, v_w_bf, v_b, Vf, B_ * TFEAT, MDIM, MDIM, MDIM);

    for (int t = 0; t < TDEC; ++t) {
        sample_embed<<<dim3(B_), dim3(256), 0, stream>>>(input, outs, emb, context, x_bf, t);
        gemm(x_bf,      w_ih_bf, b_ih, gx, B_, 3 * MDIM, 2 * MDIM, 3 * MDIM);
        gemm(hidden_bf, w_hh_bf, b_hh, gh, B_, 3 * MDIM, MDIM,     3 * MDIM);
        gru_update<<<dim3(B_ * MDIM / 256), dim3(256), 0, stream>>>(gx, gh, hidden, hidden_bf);
        gemm(hidden_bf, q_w_bf,  q_b,  qv, B_, MDIM,     MDIM,     MDIM);
        attention_kernel<<<dim3(B_), dim3(256), 0, stream>>>(qv, Kf, Vf, fmask, hidden,
                                                             context, hc_bf, attw, t);
        // logits written directly into d_out at [b, t, :] (row stride TDEC*VOC)
        gemm(hc_bf, out_w_bf, out_b, outs + (size_t)t * VOC, B_, VOC, 2 * MDIM, TDEC * VOC);
    }
}
